// Better_Transformer_38087769981049
// MI455X (gfx1250) — compile-verified, matrix operand order checked by reading, not run on hardware
//
#include <hip/hip_runtime.h>
#include <math.h>

#define B_ROWS 16384
#define F_DIM  4096
#define P_BLK  32
#define D_BLK  128
#define EPSF   1e-5f
#define MTILE  64
#define NTILES (B_ROWS / MTILE)        // 256
#define LDSB   136                     // bf16 elems per LDS row: 128 + 8 pad (272B, 16 mod 256)
#define WN     (P_BLK * D_BLK * D_BLK) // elems per weight plane

typedef __attribute__((ext_vector_type(16))) __bf16 v16bf;
typedef __attribute__((ext_vector_type(8)))  __bf16 v8bf;
typedef __attribute__((ext_vector_type(4)))  __bf16 v4bf;
typedef __attribute__((ext_vector_type(8)))  float  v8f;

// ---------- helpers ----------

static __device__ inline void split2(float x, __bf16 &h, __bf16 &l) {
  h = (__bf16)x;
  l = (__bf16)(x - (float)h);
}

static __device__ inline unsigned pack_hl(float v) {
  __bf16 h, l; split2(v, h, l);
  return ((unsigned)__builtin_bit_cast(unsigned short, l) << 16) |
         (unsigned)__builtin_bit_cast(unsigned short, h);
}

static __device__ inline v8f wmma_bf16(v16bf a, v16bf b, v8f c) {
  return __builtin_amdgcn_wmma_f32_16x16x32_bf16(
      /*neg_a=*/false, a, /*neg_b=*/false, b,
      /*c_mod=*/(short)0, c, /*reuse_a=*/false, /*reuse_b=*/false);
}

// split-bf16 emulated-f32 product: ah*bh + ah*bl + al*bh
static __device__ inline v8f mm3(v16bf ah, v16bf al, v16bf bh, v16bf bl, v8f c) {
  c = wmma_bf16(ah, bh, c);
  c = wmma_bf16(ah, bl, c);
  c = wmma_bf16(al, bh, c);
  return c;
}

// Fragment = elems [kb..kb+7] and [kb+16..kb+23] of a contiguous bf16 row.
// Two 16B loads (ds_load_b128 from LDS / global_load_b128 from memory), no repack.
static __device__ inline v16bf frag_ld(const __bf16* base, int kb) {
  v8bf a0 = *(const v8bf*)(base + kb);
  v8bf a1 = *(const v8bf*)(base + kb + 16);
  return __builtin_shufflevector(a0, a1, 0, 1, 2, 3, 4, 5, 6, 7,
                                 8, 9, 10, 11, 12, 13, 14, 15);
}

// Cooperatively stage a 64x128 f32 tile into hi/lo bf16 LDS planes.
// Each element converted exactly once.
static __device__ inline void stage_x(const float* __restrict__ src,
                                      __bf16* sh, __bf16* sl) {
  const int t  = threadIdx.x;
  const int r  = t >> 2;          // 0..63
  const int c0 = (t & 3) * 32;    // 0/32/64/96
  const float* rp = src + (size_t)r * F_DIM + c0;
  __bf16* dh = sh + r * LDSB + c0;
  __bf16* dl = sl + r * LDSB + c0;
#pragma unroll
  for (int i = 0; i < 8; i++) {
    float4 v = ((const float4*)rp)[i];
    __bf16 h0, l0, h1, l1, h2, l2, h3, l3;
    split2(v.x, h0, l0); split2(v.y, h1, l1);
    split2(v.z, h2, l2); split2(v.w, h3, l3);
    v4bf vh; vh[0] = h0; vh[1] = h1; vh[2] = h2; vh[3] = h3;
    v4bf vl; vl[0] = l0; vl[1] = l1; vl[2] = l2; vl[3] = l3;
    *(v4bf*)(dh + 4 * i) = vh;
    *(v4bf*)(dl + 4 * i) = vl;
  }
}

// ---------- kernel 0: pre-split weights into transposed hi/lo bf16 planes ----------
// out[p][e][d] = split(w[p][d][e]) so B-fragment reads are contiguous in d.

__global__ void k_prep_w(const float* __restrict__ w,
                         __bf16* __restrict__ th, __bf16* __restrict__ tl) {
  int i = blockIdx.x * 256 + threadIdx.x;     // over P*D*D
  int d = i & 127, e = (i >> 7) & 127, p = i >> 14;
  float v = w[((size_t)p << 14) | ((size_t)d << 7) | (size_t)e];
  __bf16 h, l; split2(v, h, l);
  th[i] = h; tl[i] = l;
}

// ---------- kernel 1: y1 = bmm(x,w1)+b1 (registers only) -> per-feature partials ----------

__global__ __launch_bounds__(256) void k_bmm1_stats(
    const float* __restrict__ x,
    const __bf16* __restrict__ wth, const __bf16* __restrict__ wtl,
    const float* __restrict__ bias1,
    float* __restrict__ psum, float* __restrict__ psumsq) {
  __shared__ __bf16 sh[MTILE * LDSB];
  __shared__ __bf16 sl[MTILE * LDSB];

  const int mtile = blockIdx.x;
  const int p     = blockIdx.y;
  const int row0  = mtile * MTILE;

  stage_x(x + (size_t)row0 * F_DIM + p * D_BLK, sh, sl);
  __syncthreads();

  const int wave = threadIdx.x >> 5;
  const int lane = threadIdx.x & 31;
  const int lmod = lane & 15;
  const int lhi  = lane >> 4;
  const int colb = wave * 16 + lmod;
  const int f    = p * D_BLK + colb;

  const __bf16* wbh = wth + ((size_t)p * D_BLK + colb) * D_BLK;
  const __bf16* wbl = wtl + ((size_t)p * D_BLK + colb) * D_BLK;
  v16bf bh[4], bl[4];
#pragma unroll
  for (int c = 0; c < 4; c++) {
    int kb = 32 * c + 8 * lhi;
    bh[c] = frag_ld(wbh, kb);
    bl[c] = frag_ld(wbl, kb);
  }

  const float biasv = bias1[f];
  float s = 0.f, s2 = 0.f;

#pragma unroll
  for (int m = 0; m < MTILE / 16; m++) {
    const __bf16* ab = sh + (m * 16 + lmod) * LDSB;
    const __bf16* lb = sl + (m * 16 + lmod) * LDSB;
    v8f acc;
#pragma unroll
    for (int r = 0; r < 8; r++) acc[r] = biasv;
#pragma unroll
    for (int c = 0; c < 4; c++) {
      int kb = 32 * c + 8 * lhi;
      acc = mm3(frag_ld(ab, kb), frag_ld(lb, kb), bh[c], bl[c], acc);
    }
#pragma unroll
    for (int r = 0; r < 8; r++) { float v = acc[r]; s += v; s2 += v * v; }
  }
  s  += __shfl_xor(s, 16, 32);
  s2 += __shfl_xor(s2, 16, 32);
  if (lhi == 0) {
    psum[(size_t)mtile * F_DIM + f]   = s;
    psumsq[(size_t)mtile * F_DIM + f] = s2;
  }
}

// ---------- kernel 2/4: deterministic stats finalize ----------

__global__ void k_finalize(const float* __restrict__ psum, const float* __restrict__ psumsq,
                           float* __restrict__ mean, float* __restrict__ invstd) {
  int f = blockIdx.x * blockDim.x + threadIdx.x;
  if (f >= F_DIM) return;
  float s = 0.f, s2 = 0.f;
  for (int t = 0; t < NTILES; t++) {
    s  += psum[(size_t)t * F_DIM + f];
    s2 += psumsq[(size_t)t * F_DIM + f];
  }
  const float inv_n = 1.0f / (float)B_ROWS;
  float m = s * inv_n;
  float v = s2 * inv_n - m * m;
  mean[f]   = m;
  invstd[f] = rsqrtf(v + EPSF);
}

// ---------- kernel 3: recompute y1, BN1+tanh -> o1, bmm2 + b2 + residual -> y2, stats2 ----------

__global__ __launch_bounds__(256) void k_fused_mid(
    const float* __restrict__ x,
    const __bf16* __restrict__ wt1h, const __bf16* __restrict__ wt1l,
    const float* __restrict__ bias1,
    const __bf16* __restrict__ wt2h, const __bf16* __restrict__ wt2l,
    const float* __restrict__ bias2,
    const float* __restrict__ gamma1, const float* __restrict__ beta1,
    const float* __restrict__ mean1, const float* __restrict__ invstd1,
    float* __restrict__ y2, float* __restrict__ psum, float* __restrict__ psumsq) {
  __shared__ __bf16 sh[MTILE * LDSB];   // phase1: x-hi, phase2: o1-hi
  __shared__ __bf16 sl[MTILE * LDSB];   // phase1: x-lo, phase2: o1-lo

  const int mtile = blockIdx.x;
  const int p     = blockIdx.y;
  const int row0  = mtile * MTILE;

  stage_x(x + (size_t)row0 * F_DIM + p * D_BLK, sh, sl);
  __syncthreads();

  const int wave = threadIdx.x >> 5;
  const int lane = threadIdx.x & 31;
  const int lmod = lane & 15;
  const int lhi  = lane >> 4;
  const int colb = wave * 16 + lmod;
  const int f    = p * D_BLK + colb;

  // ---- phase 1: GEMM1 + BN1 + tanh, o1 kept packed in registers ----
  unsigned opack[MTILE / 16][8];
  {
    const __bf16* wbh = wt1h + ((size_t)p * D_BLK + colb) * D_BLK;
    const __bf16* wbl = wt1l + ((size_t)p * D_BLK + colb) * D_BLK;
    v16bf bh[4], bl[4];
#pragma unroll
    for (int c = 0; c < 4; c++) {
      int kb = 32 * c + 8 * lhi;
      bh[c] = frag_ld(wbh, kb);
      bl[c] = frag_ld(wbl, kb);
    }
    const float biasv = bias1[f];
    const float g1 = gamma1[f], bt1 = beta1[f], mu1 = mean1[f], is1 = invstd1[f];

#pragma unroll
    for (int m = 0; m < MTILE / 16; m++) {
      const __bf16* ab = sh + (m * 16 + lmod) * LDSB;
      const __bf16* lb = sl + (m * 16 + lmod) * LDSB;
      v8f acc;
#pragma unroll
      for (int r = 0; r < 8; r++) acc[r] = biasv;
#pragma unroll
      for (int c = 0; c < 4; c++) {
        int kb = 32 * c + 8 * lhi;
        acc = mm3(frag_ld(ab, kb), frag_ld(lb, kb), bh[c], bl[c], acc);
      }
#pragma unroll
      for (int r = 0; r < 8; r++)
        opack[m][r] = pack_hl(tanhf(g1 * (acc[r] - mu1) * is1 + bt1));
    }
  }
  __syncthreads();   // everyone done reading x planes

  // ---- write o1 hi/lo planes (reuse LDS) ----
#pragma unroll
  for (int m = 0; m < MTILE / 16; m++) {
#pragma unroll
    for (int r = 0; r < 8; r++) {
      int rl = m * 16 + r + 8 * lhi;
      unsigned u = opack[m][r];
      sh[rl * LDSB + colb] = __builtin_bit_cast(__bf16, (unsigned short)(u & 0xffffu));
      sl[rl * LDSB + colb] = __builtin_bit_cast(__bf16, (unsigned short)(u >> 16));
    }
  }
  __syncthreads();

  // ---- phase 2: y2 = o1 @ w2 + b2 + x ----
  {
    const __bf16* wbh = wt2h + ((size_t)p * D_BLK + colb) * D_BLK;
    const __bf16* wbl = wt2l + ((size_t)p * D_BLK + colb) * D_BLK;
    v16bf bh[4], bl[4];
#pragma unroll
    for (int c = 0; c < 4; c++) {
      int kb = 32 * c + 8 * lhi;
      bh[c] = frag_ld(wbh, kb);
      bl[c] = frag_ld(wbl, kb);
    }
    const float biasv = bias2[f];
    float s = 0.f, s2 = 0.f;

#pragma unroll
    for (int m = 0; m < MTILE / 16; m++) {
      const __bf16* ab = sh + (m * 16 + lmod) * LDSB;
      const __bf16* lb = sl + (m * 16 + lmod) * LDSB;
      v8f acc;
#pragma unroll
      for (int r = 0; r < 8; r++) acc[r] = biasv;
#pragma unroll
      for (int c = 0; c < 4; c++) {
        int kb = 32 * c + 8 * lhi;
        acc = mm3(frag_ld(ab, kb), frag_ld(lb, kb), bh[c], bl[c], acc);
      }
#pragma unroll
      for (int r = 0; r < 8; r++) {
        size_t row = (size_t)(row0 + m * 16 + r + 8 * lhi);
        float v = acc[r] + x[row * F_DIM + f];   // residual (L2-hot tile)
        y2[row * F_DIM + f] = v;
        s += v; s2 += v * v;
      }
    }
    s  += __shfl_xor(s, 16, 32);
    s2 += __shfl_xor(s2, 16, 32);
    if (lhi == 0) {
      psum[(size_t)mtile * F_DIM + f]   = s;
      psumsq[(size_t)mtile * F_DIM + f] = s2;
    }
  }
}

// ---------- kernel 5: out = tanh(BN2(y2)) in place, float4 ----------

__global__ void k_bn_tanh(float* __restrict__ y,
                          const float* __restrict__ gamma, const float* __restrict__ beta,
                          const float* __restrict__ mean, const float* __restrict__ invstd) {
  size_t i = (size_t)blockIdx.x * blockDim.x + threadIdx.x;
  const size_t n4 = (size_t)B_ROWS * F_DIM / 4;
  if (i >= n4) return;
  float4 v = ((const float4*)y)[i];
  int f0 = (int)((i * 4) & (F_DIM - 1));
  v.x = tanhf(gamma[f0 + 0] * (v.x - mean[f0 + 0]) * invstd[f0 + 0] + beta[f0 + 0]);
  v.y = tanhf(gamma[f0 + 1] * (v.y - mean[f0 + 1]) * invstd[f0 + 1] + beta[f0 + 1]);
  v.z = tanhf(gamma[f0 + 2] * (v.z - mean[f0 + 2]) * invstd[f0 + 2] + beta[f0 + 2]);
  v.w = tanhf(gamma[f0 + 3] * (v.w - mean[f0 + 3]) * invstd[f0 + 3] + beta[f0 + 3]);
  ((float4*)y)[i] = v;
}

// ---------- launch ----------

extern "C" void kernel_launch(void* const* d_in, const int* in_sizes, int n_in,
                              void* d_out, int out_size, void* d_ws, size_t ws_size,
                              hipStream_t stream) {
  const float* x      = (const float*)d_in[0];
  const float* w1     = (const float*)d_in[1];
  const float* bias1  = (const float*)d_in[2];
  const float* w2     = (const float*)d_in[3];
  const float* bias2  = (const float*)d_in[4];
  const float* gamma1 = (const float*)d_in[5];
  const float* beta1  = (const float*)d_in[6];
  const float* gamma3 = (const float*)d_in[7];
  const float* beta3  = (const float*)d_in[8];
  float* out = (float*)d_out;

  const size_t NP = (size_t)NTILES * F_DIM;
  float* ws    = (float*)d_ws;
  float* psum1 = ws;
  float* psq1  = psum1 + NP;
  float* psum2 = psq1 + NP;
  float* psq2  = psum2 + NP;
  float* mean1 = psq2 + NP;
  float* istd1 = mean1 + F_DIM;
  float* mean2 = istd1 + F_DIM;
  float* istd2 = mean2 + F_DIM;
  __bf16* wt1h = (__bf16*)(istd2 + F_DIM);
  __bf16* wt1l = wt1h + WN;
  __bf16* wt2h = wt1l + WN;
  __bf16* wt2l = wt2h + WN;

  dim3 gt(NTILES, P_BLK), bt(256);

  k_prep_w<<<WN / 256, 256, 0, stream>>>(w1, wt1h, wt1l);
  k_prep_w<<<WN / 256, 256, 0, stream>>>(w2, wt2h, wt2l);
  k_bmm1_stats<<<gt, bt, 0, stream>>>(x, wt1h, wt1l, bias1, psum1, psq1);
  k_finalize<<<F_DIM / 256, 256, 0, stream>>>(psum1, psq1, mean1, istd1);
  k_fused_mid<<<gt, bt, 0, stream>>>(x, wt1h, wt1l, bias1, wt2h, wt2l, bias2,
                                     gamma1, beta1, mean1, istd1,
                                     out, psum2, psq2);
  k_finalize<<<F_DIM / 256, 256, 0, stream>>>(psum2, psq2, mean2, istd2);
  k_bn_tanh<<<(unsigned)((size_t)B_ROWS * F_DIM / 4 / 256), 256, 0, stream>>>(
      out, gamma3, beta3, mean2, istd2);
}
